// CRFBaseModel_81217831568011
// MI455X (gfx1250) — compile-verified, matrix operand order checked by reading, not run on hardware
//
#include <hip/hip_runtime.h>
#include <hip/hip_bf16.h>

#define NSTATE 101
#define MAXLEN 100
#define SEQ_L  1024
#define BATCH  256
#define FEAT_D 64
#define NEGV   (-10000.0f)
#define CHUNK  256
#define NCHNK  (SEQ_L / CHUNK)
#define CHUNK_DW (CHUNK * 3)

typedef __attribute__((ext_vector_type(16))) _Float16 v16h;
typedef __attribute__((ext_vector_type(8)))  float    v8f;
typedef unsigned int tdm_v4u __attribute__((ext_vector_type(4)));
typedef int          tdm_v8i __attribute__((ext_vector_type(8)));
typedef int          tdm_v4i __attribute__((ext_vector_type(4)));

#if __has_builtin(__builtin_amdgcn_tensor_load_to_lds)
#define USE_TDM 1
#else
#define USE_TDM 0
#endif

// ---------- helpers ----------
__device__ __forceinline__ float lse2f(float a, float b) {
    float m = fmaxf(a, b);
    return m + __logf(__expf(a - m) + __expf(b - m));
}
__device__ __forceinline__ float endc(int j, const float* __restrict__ end_trans) {
    return (j == 0 || j == MAXLEN) ? end_trans[j] : NEGV;
}
__device__ __forceinline__ float transc(int a, int c, const float* __restrict__ trans) {
    bool allowed = (a == 0   && (c == 0 || c == 1))
                || (a >= 1   && a <= 98 && c == a + 1)
                || (a >= 4   && a <= 99 && c == 100)
                || (a == 99  && c == 99)
                || (a == 100 && (c == 0 || c == 1));
    return allowed ? trans[a * NSTATE + c] : NEGV;
}

// ---------- Tensor Data Mover staging of emission chunks ----------
__device__ __forceinline__ void tdm_issue(const float* gsrc, float* ldst, int ndw)
{
#if USE_TDM
    unsigned long long ga = (unsigned long long)gsrc;
    tdm_v4u g0;
    g0[0] = 1u;                                       // count=1, user descriptor
    g0[1] = (unsigned)(unsigned long long)ldst;       // lds_addr (low 32 bits of flat = LDS offset)
    g0[2] = (unsigned)ga;                             // global_addr[31:0]
    g0[3] = (unsigned)(ga >> 32) | 0x80000000u;       // global_addr[56:32] | type=2
    tdm_v8i g1;
    g1[0] = 0x00020000;                               // data_size=2 (4B), no multicast/pad
    g1[1] = (ndw & 0xFFFF) << 16;                     // tensor_dim0[15:0]
    g1[2] = (int)(((unsigned)ndw >> 16) | (1u << 16));// tensor_dim0[31:16], tensor_dim1=1
    g1[3] = (ndw & 0xFFFF) << 16;                     // tile_dim0 = ndw
    g1[4] = 1;                                        // tile_dim1=1, tile_dim2=0
    g1[5] = ndw;                                      // tensor_dim0_stride[31:0]
    g1[6] = 0;
    g1[7] = 0;
    tdm_v4i z4 = {0, 0, 0, 0};
#if defined(__clang_major__) && __clang_major__ >= 23
    tdm_v8i z8 = {0, 0, 0, 0, 0, 0, 0, 0};
    __builtin_amdgcn_tensor_load_to_lds(g0, g1, z4, z4, z8, 0);
#else
    __builtin_amdgcn_tensor_load_to_lds(g0, g1, z4, z4, 0);
#endif
#else
    (void)gsrc; (void)ldst; (void)ndw;
#endif
}

__device__ __forceinline__ void stage_chunk(const float* gsrc, float* ldst, int wid, int j)
{
#if USE_TDM
    if (wid == 0) tdm_issue(gsrc, ldst, CHUNK_DW);    // wave-uniform branch; TDM ignores EXEC
#else
    (void)wid;
    for (int k = j; k < CHUNK_DW; k += 128) ldst[k] = gsrc[k];
#endif
}

__device__ __forceinline__ void wait_chunk(int wid)
{
#if USE_TDM
    if (wid == 0) __builtin_amdgcn_s_wait_tensorcnt(0);
#else
    (void)wid;
#endif
    __syncthreads();
}

// ======================================================================
// 1) Emissions via WMMA: E(16x16 tile) = F(16x64) x Wpad(64x16) + bias
// ======================================================================
__global__ void __launch_bounds__(256)
emission_wmma_kernel(const float* __restrict__ feat,   // (B*L, 64)
                     const float* __restrict__ W,      // (64, 3)
                     const float* __restrict__ bias,   // (3)
                     float* __restrict__ em3)          // (B*L, 3)
{
    const int lane = threadIdx.x & 31;
    const int wid  = threadIdx.x >> 5;
    const int hi   = lane >> 4;
    const int lo   = lane & 15;
    const long rowBase = (long)blockIdx.x * 128 + (long)wid * 16;

    v16h b0, b1;
    const int n = lo;
#pragma unroll
    for (int h = 0; h < 16; ++h) {
        int k = (h < 8) ? (8 * hi + h) : (8 + 8 * hi + h);
        float w0 = (n < 3) ? W[k * 3 + n]        : 0.0f;
        float w1 = (n < 3) ? W[(k + 32) * 3 + n] : 0.0f;
        b0[h] = (_Float16)w0;
        b1[h] = (_Float16)w1;
    }
    const float* frow = feat + (rowBase + lo) * FEAT_D;
    v16h a0, a1;
#pragma unroll
    for (int h = 0; h < 16; ++h) {
        int k = (h < 8) ? (8 * hi + h) : (8 + 8 * hi + h);
        a0[h] = (_Float16)frow[k];
        a1[h] = (_Float16)frow[k + 32];
    }
    float cb = 0.0f;
    if      (n == 0) cb = bias[0];
    else if (n == 1) cb = bias[1] + 0.5f;
    else if (n == 2) cb = bias[2] + 5.0f;
    v8f c;
#pragma unroll
    for (int g = 0; g < 8; ++g) c[g] = cb;

    c = __builtin_amdgcn_wmma_f32_16x16x32_f16(false, a0, false, b0, (short)0, c, false, false);
    c = __builtin_amdgcn_wmma_f32_16x16x32_f16(false, a1, false, b1, (short)0, c, false, false);

    if (n < 3) {
#pragma unroll
        for (int g = 0; g < 8; ++g) {
            long r = rowBase + g + 8 * hi;
            em3[r * 3 + n] = c[g];
        }
    }
}

// ======================================================================
// 2) Merged scans: gridDim = (BATCH, 3); y==0 alpha, y==1 beta, y==2 viterbi
// ======================================================================
__device__ void alpha_pass(int b, int j,
                           const float* __restrict__ em3,
                           const unsigned char* __restrict__ mask,
                           const float* __restrict__ trans,
                           const float* __restrict__ start_trans,
                           float* __restrict__ alpha_ws,
                           float (*sbuf)[NSTATE], float* s_redA, float* s_redB,
                           float (*em_sh)[CHUNK_DW])
{
    const int lane = j & 31, wid = j >> 5;
    float tin = NEGV;
    if (j >= 1 && j <= 99) tin = trans[(j - 1) * NSTATE + j];
    const float T00 = trans[0], T01 = trans[1];
    const float T100_0 = trans[100 * NSTATE + 0], T100_1 = trans[100 * NSTATE + 1];
    const float T9999  = trans[99 * NSTATE + 99];
    float ti100 = NEGV;
    if (j >= 4 && j <= 99) ti100 = trans[j * NSTATE + 100];

    const long base_bt = (long)b * SEQ_L;
    stage_chunk(em3 + base_bt * 3, em_sh[0], wid, j);

    float areg = NEGV;
    int p = 0;
    for (int c = 0; c < NCHNK; ++c) {
        wait_chunk(wid);                                  // chunk c resident in LDS
        if (c + 1 < NCHNK)
            stage_chunk(em3 + (base_bt + (long)(c + 1) * CHUNK) * 3,
                        em_sh[(c + 1) & 1], wid, j);
        const float* eb = em_sh[c & 1];
        const int t0 = c * CHUNK;
        for (int tt = 0; tt < CHUNK; ++tt) {
            const int t = t0 + tt;
            const float e0 = eb[tt * 3], e1 = eb[tt * 3 + 1], e2 = eb[tt * 3 + 2];
            if (t == 0) {
                if (j < NSTATE) {
                    float e = (j == 0) ? e0 : ((j == MAXLEN) ? e2 : e1);
                    areg = start_trans[j] + e;
                    sbuf[p][j] = areg;
                    alpha_ws[base_bt * NSTATE + j] = areg;
                }
                __syncthreads();
                continue;
            }
            // 96-way logsumexp into state 100
            float x = (j >= 4 && j <= 99) ? (areg + ti100) : NEGV;
            float m = x;
#pragma unroll
            for (int off = 16; off > 0; off >>= 1) m = fmaxf(m, __shfl_xor(m, off));
            float s = __expf(x - m);
#pragma unroll
            for (int off = 16; off > 0; off >>= 1) s += __shfl_xor(s, off);
            if (lane == 0) { s_redA[wid] = m; s_redB[wid] = s; }
            __syncthreads();
            float M = fmaxf(fmaxf(s_redA[0], s_redA[1]), fmaxf(s_redA[2], s_redA[3]));
            float Ssum = s_redB[0] * __expf(s_redA[0] - M) + s_redB[1] * __expf(s_redA[1] - M)
                       + s_redB[2] * __expf(s_redA[2] - M) + s_redB[3] * __expf(s_redA[3] - M);
            float lse100 = M + __logf(Ssum);

            const float* a_old = sbuf[p];
            long idx = base_bt + t;
            float newv = NEGV;
            if      (j == 0)   newv = lse2f(a_old[0] + T00, a_old[100] + T100_0) + e0;
            else if (j == 1)   newv = lse2f(a_old[0] + T01, a_old[100] + T100_1) + e1;
            else if (j <= 98)  newv = a_old[j - 1] + tin + e1;
            else if (j == 99)  newv = lse2f(a_old[98] + tin, a_old[99] + T9999) + e1;
            else if (j == 100) newv = lse100 + e2;
            const bool mt = mask[idx] != 0;
            float cur = (j < NSTATE) ? (mt ? newv : areg) : NEGV;
            if (j < NSTATE) { sbuf[p ^ 1][j] = cur; alpha_ws[idx * NSTATE + j] = cur; }
            areg = cur;
            p ^= 1;
            __syncthreads();
        }
    }
}

__device__ void beta_pass(int b, int i,
                          const float* __restrict__ em3,
                          const unsigned char* __restrict__ mask,
                          const float* __restrict__ trans,
                          const float* __restrict__ end_trans,
                          float* __restrict__ beta_ws,
                          float (*xbuf)[NSTATE],
                          float (*em_sh)[CHUNK_DW])
{
    const int wid = i >> 5;
    int s1 = -1, s2 = -1; float T1 = NEGV, T2 = NEGV;
    if      (i == 0)            { s1 = 0; T1 = trans[0]; s2 = 1; T2 = trans[1]; }
    else if (i >= 1 && i <= 3)  { s1 = i + 1; T1 = trans[i * NSTATE + i + 1]; }
    else if (i >= 4 && i <= 98) { s1 = i + 1; T1 = trans[i * NSTATE + i + 1];
                                  s2 = 100;  T2 = trans[i * NSTATE + 100]; }
    else if (i == 99)           { s1 = 99;  T1 = trans[99 * NSTATE + 99];
                                  s2 = 100; T2 = trans[99 * NSTATE + 100]; }
    else if (i == 100)          { s1 = 0; T1 = trans[100 * NSTATE + 0];
                                  s2 = 1; T2 = trans[100 * NSTATE + 1]; }

    const long base_bt = (long)b * SEQ_L;
    float breg = (i < NSTATE) ? endc(i, end_trans) : NEGV;
    if (i < NSTATE) beta_ws[(base_bt + SEQ_L - 1) * NSTATE + i] = breg;

    stage_chunk(em3 + (base_bt + (long)(NCHNK - 1) * CHUNK) * 3,
                em_sh[(NCHNK - 1) & 1], wid, i);
    int q = 0;
    for (int c = NCHNK - 1; c >= 0; --c) {
        wait_chunk(wid);
        if (c - 1 >= 0)
            stage_chunk(em3 + (base_bt + (long)(c - 1) * CHUNK) * 3,
                        em_sh[(c - 1) & 1], wid, i);
        const float* eb = em_sh[c & 1];
        const int nhi = c * CHUNK + CHUNK - 1;
        const int nlo = (c == 0) ? 1 : c * CHUNK;
        for (int n = nhi; n >= nlo; --n) {       // n = t+1 (emission index)
            const int tt = n - c * CHUNK;
            const float e0 = eb[tt * 3], e1 = eb[tt * 3 + 1], e2 = eb[tt * 3 + 2];
            float e = (i == 0) ? e0 : ((i == MAXLEN) ? e2 : e1);
            if (i < NSTATE) xbuf[q][i] = e + breg;        // x[j] = e[n][j] + beta[n][j]
            __syncthreads();
            float newv = NEGV;
            if (s1 >= 0) newv = T1 + xbuf[q][s1];
            if (s2 >= 0) newv = lse2f(newv, T2 + xbuf[q][s2]);
            const bool mn = mask[base_bt + n] != 0;
            float cur = (i < NSTATE) ? (mn ? newv : breg) : NEGV;
            if (i < NSTATE) beta_ws[(base_bt + n - 1) * NSTATE + i] = cur;
            breg = cur;
            q ^= 1;
        }
    }
}

__device__ void viterbi_pass(int b, int j,
                             const float* __restrict__ em3,
                             const unsigned char* __restrict__ mask,
                             const float* __restrict__ trans,
                             const float* __restrict__ start_trans,
                             const float* __restrict__ end_trans,
                             unsigned char* __restrict__ bp_ws,
                             int* __restrict__ last_ws,
                             float (*sbuf)[NSTATE], float* s_redA, int* s_redI,
                             float (*em_sh)[CHUNK_DW])
{
    const int lane = j & 31, wid = j >> 5;
    float tin = NEGV;
    if (j >= 1 && j <= 99) tin = trans[(j - 1) * NSTATE + j];
    const float T00 = trans[0], T01 = trans[1];
    const float T100_0 = trans[100 * NSTATE + 0], T100_1 = trans[100 * NSTATE + 1];
    const float T9999  = trans[99 * NSTATE + 99];
    float ti100 = NEGV;
    if (j >= 4 && j <= 99) ti100 = trans[j * NSTATE + 100];

    const long base_bt = (long)b * SEQ_L;
    stage_chunk(em3 + base_bt * 3, em_sh[0], wid, j);

    float vreg = NEGV;
    int p = 0;
    for (int c = 0; c < NCHNK; ++c) {
        wait_chunk(wid);
        if (c + 1 < NCHNK)
            stage_chunk(em3 + (base_bt + (long)(c + 1) * CHUNK) * 3,
                        em_sh[(c + 1) & 1], wid, j);
        const float* eb = em_sh[c & 1];
        const int t0 = c * CHUNK;
        for (int tt = 0; tt < CHUNK; ++tt) {
            const int t = t0 + tt;
            const float e0 = eb[tt * 3], e1 = eb[tt * 3 + 1], e2 = eb[tt * 3 + 2];
            if (t == 0) {
                if (j < NSTATE) {
                    float e = (j == 0) ? e0 : ((j == MAXLEN) ? e2 : e1);
                    vreg = start_trans[j] + e;
                    sbuf[p][j] = vreg;
                }
                __syncthreads();
                continue;
            }
            // argmax over i=4..99 of v[i]+T[i,100], ties -> smallest index
            float x = (j >= 4 && j <= 99) ? (vreg + ti100) : NEGV;
            float m = x; int mi = (j < NSTATE) ? j : 127;
#pragma unroll
            for (int off = 16; off > 0; off >>= 1) {
                float om = __shfl_xor(m, off); int oi = __shfl_xor(mi, off);
                if (om > m || (om == m && oi < mi)) { m = om; mi = oi; }
            }
            if (lane == 0) { s_redA[wid] = m; s_redI[wid] = mi; }
            __syncthreads();
            float M = s_redA[0]; int MI = s_redI[0];
#pragma unroll
            for (int w = 1; w < 4; ++w)
                if (s_redA[w] > M || (s_redA[w] == M && s_redI[w] < MI)) { M = s_redA[w]; MI = s_redI[w]; }

            const float* v_old = sbuf[p];
            long idx = base_bt + t;
            float newv = NEGV; int bp = j;
            if (j == 0) {
                float c0 = v_old[0] + T00, c1 = v_old[100] + T100_0;
                newv = fmaxf(c0, c1) + e0; bp = (c0 >= c1) ? 0 : 100;
            } else if (j == 1) {
                float c0 = v_old[0] + T01, c1 = v_old[100] + T100_1;
                newv = fmaxf(c0, c1) + e1; bp = (c0 >= c1) ? 0 : 100;
            } else if (j <= 98) {
                newv = v_old[j - 1] + tin + e1; bp = j - 1;
            } else if (j == 99) {
                float c0 = v_old[98] + tin, c1 = v_old[99] + T9999;
                newv = fmaxf(c0, c1) + e1; bp = (c0 >= c1) ? 98 : 99;
            } else if (j == 100) {
                newv = M + e2; bp = MI;
            }
            const bool mt = mask[idx] != 0;
            float cur = NEGV; int bpw = j;
            if (j < NSTATE) { cur = mt ? newv : vreg; bpw = mt ? bp : j; }
            if (j < NSTATE) {
                sbuf[p ^ 1][j] = cur;
                bp_ws[idx * NSTATE + j] = (unsigned char)bpw;
            }
            vreg = cur;
            p ^= 1;
            __syncthreads();
        }
    }
    if (j < NSTATE) sbuf[0][j] = vreg + endc(j, end_trans);
    __syncthreads();
    if (j == 0) {
        float best = sbuf[0][0]; int bi = 0;
        for (int k = 1; k < NSTATE; ++k)
            if (sbuf[0][k] > best) { best = sbuf[0][k]; bi = k; }
        last_ws[b] = bi;
    }
}

__global__ void __launch_bounds__(128)
scan3_kernel(const float* __restrict__ em3,
             const unsigned char* __restrict__ mask,
             const float* __restrict__ trans,
             const float* __restrict__ start_trans,
             const float* __restrict__ end_trans,
             float* __restrict__ alpha_ws,
             float* __restrict__ beta_ws,
             unsigned char* __restrict__ bp_ws,
             int* __restrict__ last_ws)
{
    __shared__ float s_buf[2][NSTATE];
    __shared__ float s_redA[4];
    __shared__ float s_redB[4];
    __shared__ int   s_redI[4];
    __shared__ float em_sh[2][CHUNK_DW];
    const int b = blockIdx.x;
    const int j = threadIdx.x;
    if (blockIdx.y == 0)
        alpha_pass(b, j, em3, mask, trans, start_trans, alpha_ws, s_buf, s_redA, s_redB, em_sh);
    else if (blockIdx.y == 1)
        beta_pass(b, j, em3, mask, trans, end_trans, beta_ws, s_buf, em_sh);
    else
        viterbi_pass(b, j, em3, mask, trans, start_trans, end_trans, bp_ws, last_ws,
                     s_buf, s_redA, s_redI, em_sh);
}

// ======================================================================
// 3) probs = softmax(alpha + beta) over states, one (b,t) row per block
// ======================================================================
__global__ void __launch_bounds__(128)
probs_kernel(const float* __restrict__ alpha_ws,
             const float* __restrict__ beta_ws,
             float* __restrict__ probs_out)
{
    const long row = blockIdx.x;
    const int j = threadIdx.x;
    const int lane = j & 31, wid = j >> 5;
    __shared__ float m_sh[4], s_sh[4];
    float x = NEGV;
    if (j < NSTATE) x = alpha_ws[row * NSTATE + j] + beta_ws[row * NSTATE + j];
    float m = x;
#pragma unroll
    for (int off = 16; off > 0; off >>= 1) m = fmaxf(m, __shfl_xor(m, off));
    if (lane == 0) m_sh[wid] = m;
    __syncthreads();
    float M = fmaxf(fmaxf(m_sh[0], m_sh[1]), fmaxf(m_sh[2], m_sh[3]));
    float ex = __expf(x - M);
    float s = ex;
#pragma unroll
    for (int off = 16; off > 0; off >>= 1) s += __shfl_xor(s, off);
    if (lane == 0) s_sh[wid] = s;
    __syncthreads();
    float Ssum = s_sh[0] + s_sh[1] + s_sh[2] + s_sh[3];
    if (j < NSTATE) probs_out[row * NSTATE + j] = ex / Ssum;
}

// ======================================================================
// 4) Viterbi backtrack: one lane per batch element, serial bp chase
// ======================================================================
__global__ void __launch_bounds__(256)
backtrack_kernel(const unsigned char* __restrict__ bp_ws,
                 const int* __restrict__ last_ws,
                 float* __restrict__ path_out)
{
    const int b = threadIdx.x;
    const long base_bt = (long)b * SEQ_L;
    int s = last_ws[b];
    path_out[base_bt + SEQ_L - 1] = (float)s;
    for (int t = SEQ_L - 1; t >= 1; --t) {
        s = bp_ws[(base_bt + t) * NSTATE + s];
        path_out[base_bt + t - 1] = (float)s;
    }
}

// ======================================================================
// 5) Loss: per-batch (score - logZ), then mean-reduce
// ======================================================================
__global__ void __launch_bounds__(128)
loss_perb_kernel(const float* __restrict__ em3,
                 const unsigned char* __restrict__ mask,
                 const int* __restrict__ targets,
                 const float* __restrict__ trans,
                 const float* __restrict__ start_trans,
                 const float* __restrict__ end_trans,
                 const float* __restrict__ alpha_ws,
                 float* __restrict__ perb_ws)
{
    const int b = blockIdx.x;
    const int j = threadIdx.x;
    const int lane = j & 31, wid = j >> 5;
    __shared__ float r_sh[4];
    __shared__ float a_sh[4];
    __shared__ int   m_sh[4];
    const long base_bt = (long)b * SEQ_L;

    float x = NEGV;
    if (j < NSTATE) x = alpha_ws[(base_bt + SEQ_L - 1) * NSTATE + j] + endc(j, end_trans);
    float m = x;
#pragma unroll
    for (int off = 16; off > 0; off >>= 1) m = fmaxf(m, __shfl_xor(m, off));
    if (lane == 0) r_sh[wid] = m;
    __syncthreads();
    float M = fmaxf(fmaxf(r_sh[0], r_sh[1]), fmaxf(r_sh[2], r_sh[3]));
    __syncthreads();
    float s = __expf(x - M);
#pragma unroll
    for (int off = 16; off > 0; off >>= 1) s += __shfl_xor(s, off);
    if (lane == 0) r_sh[wid] = s;
    __syncthreads();
    float logZ = M + __logf(r_sh[0] + r_sh[1] + r_sh[2] + r_sh[3]);

    float acc = 0.0f;
    int msum = 0;
    for (int t = j; t < SEQ_L; t += 128) {
        long idx = base_bt + t;
        if (t + 128 < SEQ_L) __builtin_prefetch(&em3[(idx + 128) * 3], 0, 0);
        int tg = targets[idx];
        bool mt = mask[idx] != 0;
        float mf = mt ? 1.0f : 0.0f;
        int col = (tg == 0) ? 0 : ((tg == MAXLEN) ? 2 : 1);
        acc += mf * em3[idx * 3 + col];
        if (t >= 1) acc += mf * transc(targets[idx - 1], tg, trans);
        msum += mt ? 1 : 0;
    }
    float av = acc;
#pragma unroll
    for (int off = 16; off > 0; off >>= 1) av += __shfl_xor(av, off);
    int mv = msum;
#pragma unroll
    for (int off = 16; off > 0; off >>= 1) mv += __shfl_xor(mv, off);
    if (lane == 0) { a_sh[wid] = av; m_sh[wid] = mv; }
    __syncthreads();
    if (j == 0) {
        float acc_t = a_sh[0] + a_sh[1] + a_sh[2] + a_sh[3];
        int msum_t  = m_sh[0] + m_sh[1] + m_sh[2] + m_sh[3];
        int last_idx = msum_t - 1;
        int last_tag = targets[base_bt + last_idx];
        int tg0 = targets[base_bt];
        float score = start_trans[tg0] + acc_t + endc(last_tag, end_trans);
        perb_ws[b] = score - logZ;
    }
}

__global__ void __launch_bounds__(256)
loss_final_kernel(const float* __restrict__ perb_ws, float* __restrict__ loss_out)
{
    __shared__ float p_sh[8];
    const int j = threadIdx.x, lane = j & 31, wid = j >> 5;
    float v = perb_ws[j];
#pragma unroll
    for (int off = 16; off > 0; off >>= 1) v += __shfl_xor(v, off);
    if (lane == 0) p_sh[wid] = v;
    __syncthreads();
    if (j == 0) {
        float t = 0.0f;
        for (int w = 0; w < 8; ++w) t += p_sh[w];
        loss_out[0] = -t / (float)BATCH;
    }
}

// ======================================================================
extern "C" void kernel_launch(void* const* d_in, const int* in_sizes, int n_in,
                              void* d_out, int out_size, void* d_ws, size_t ws_size,
                              hipStream_t stream)
{
    (void)in_sizes; (void)n_in; (void)out_size; (void)ws_size;
    const float*         features    = (const float*)d_in[0];
    const unsigned char* mask        = (const unsigned char*)d_in[1];
    const int*           targets     = (const int*)d_in[2];
    const float*         W           = (const float*)d_in[3];
    const float*         bias        = (const float*)d_in[4];
    const float*         trans       = (const float*)d_in[5];
    const float*         start_trans = (const float*)d_in[6];
    const float*         end_trans   = (const float*)d_in[7];

    const long BL = (long)BATCH * SEQ_L;
    float* out       = (float*)d_out;
    float* probs_out = out;
    float* path_out  = out + BL * NSTATE;
    float* loss_out  = path_out + BL;

    char* ws = (char*)d_ws;
    size_t off = 0;
    float* em3      = (float*)(ws + off); off += (size_t)BL * 3 * sizeof(float);
    float* alpha_ws = (float*)(ws + off); off += (size_t)BL * NSTATE * sizeof(float);
    float* beta_ws  = (float*)(ws + off); off += (size_t)BL * NSTATE * sizeof(float);
    unsigned char* bp_ws = (unsigned char*)(ws + off); off += (size_t)BL * NSTATE;
    off = (off + 255) & ~(size_t)255;
    int*   last_ws  = (int*)(ws + off);   off += BATCH * sizeof(int);
    float* perb_ws  = (float*)(ws + off); off += BATCH * sizeof(float);

    emission_wmma_kernel<<<(int)(BL / 128), 256, 0, stream>>>(features, W, bias, em3);
    scan3_kernel<<<dim3(BATCH, 3), 128, 0, stream>>>(em3, mask, trans, start_trans, end_trans,
                                                     alpha_ws, beta_ws, bp_ws, last_ws);
    probs_kernel<<<(int)BL, 128, 0, stream>>>(alpha_ws, beta_ws, probs_out);
    backtrack_kernel<<<1, 256, 0, stream>>>(bp_ws, last_ws, path_out);
    loss_perb_kernel<<<BATCH, 128, 0, stream>>>(em3, mask, targets, trans, start_trans, end_trans,
                                                alpha_ws, perb_ws);
    loss_final_kernel<<<1, 256, 0, stream>>>(perb_ws, loss_out);
}
